// QuantumKernelMethod_65481071405830
// MI455X (gfx1250) — matrix operand order repaired and checked by reading, not compiled, not run on hardware
//
#include <hip/hip_runtime.h>

// ---------------------------------------------------------------------------
// RBF kernel K[i,j] = exp(-gamma * max(||x_i||^2 + ||y_j||^2 - 2 x_i.y_j, 0))
// B = 4096, F = 1024, fp32 in/out.
//
// Pipeline:
//   1) bf16_split_kernel: x,y (fp32) -> hi/lo bf16 arrays in workspace (once).
//   2) rbf_row_norms:     squared row norms -> workspace.
//   3) rbf_wmma_gemm:     x@y^T on v_wmma_f32_16x16x32_bf16 (hi*hi+hi*lo+lo*hi,
//                         ~fp32 accuracy) + fused RBF epilogue, NT stores.
//
// GEMM: 64x64 wave tile (4x4 WMMA tiles), 48 WMMAs per K=32 step against
// 32 b128 fragment loads (10.7 B/WMMA/lane) -- register-reuse limited, not
// cache-bandwidth limited. ~250 VGPRs -> 2 waves/SIMD resident.
// ---------------------------------------------------------------------------

#define B_DIM 4096
#define F_DIM 1024

typedef __attribute__((ext_vector_type(16))) __bf16 v16bf;
typedef __attribute__((ext_vector_type(8)))  __bf16 v8bf;
typedef __attribute__((ext_vector_type(8)))  float  v8f;

// ---- fp32 -> bf16 (RNE) bit tricks ----------------------------------------
__device__ __forceinline__ unsigned short f2bf_bits(float f) {
    unsigned u = __float_as_uint(f);
    unsigned r = u + 0x7fffu + ((u >> 16) & 1u);
    return (unsigned short)(r >> 16);
}

__device__ __forceinline__ void bf_split(float f, unsigned short& hb, unsigned short& lb) {
    unsigned u = __float_as_uint(f);
    unsigned r = (u + 0x7fffu + ((u >> 16) & 1u)) & 0xffff0000u;
    hb = (unsigned short)(r >> 16);
    lb = f2bf_bits(f - __uint_as_float(r));
}

// ---------------------------------------------------------------------------
// Kernel 1: fp32 -> (hi, lo) bf16 split, 4 elements per thread.
// ---------------------------------------------------------------------------
__global__ __launch_bounds__(256)
void bf16_split_kernel(const float* __restrict__ src,
                       __bf16* __restrict__ hi, __bf16* __restrict__ lo, int n4) {
    int i = (int)(blockIdx.x * blockDim.x + threadIdx.x);
    if (i >= n4) return;
    float4 f = ((const float4*)src)[i];
    float t[4] = {f.x, f.y, f.z, f.w};
    unsigned short hb[4], lb[4];
    #pragma unroll
    for (int j = 0; j < 4; ++j) bf_split(t[j], hb[j], lb[j]);
    uint2 hv, lv;
    hv.x = (unsigned)hb[0] | ((unsigned)hb[1] << 16);
    hv.y = (unsigned)hb[2] | ((unsigned)hb[3] << 16);
    lv.x = (unsigned)lb[0] | ((unsigned)lb[1] << 16);
    lv.y = (unsigned)lb[2] | ((unsigned)lb[3] << 16);
    ((uint2*)hi)[i] = hv;
    ((uint2*)lo)[i] = lv;
}

// ---------------------------------------------------------------------------
// Kernel 2: squared row norms; one wave32 per row. x rows then y rows.
// ---------------------------------------------------------------------------
__global__ __launch_bounds__(256)
void rbf_row_norms(const float* __restrict__ x, const float* __restrict__ y,
                   float* __restrict__ sq) {
    const int gwave = (int)((blockIdx.x * blockDim.x + threadIdx.x) >> 5);
    const int lane  = (int)(threadIdx.x & 31u);
    if (gwave >= 2 * B_DIM) return;
    const float* src = (gwave < B_DIM)
                         ? (x + (size_t)gwave * F_DIM)
                         : (y + (size_t)(gwave - B_DIM) * F_DIM);
    float s = 0.0f;
    #pragma unroll 4
    for (int i = lane; i < F_DIM; i += 32) {
        float v = src[i];
        s = fmaf(v, v, s);
    }
    #pragma unroll
    for (int off = 16; off > 0; off >>= 1)
        s += __shfl_xor(s, off, 32);
    if (lane == 0) sq[gwave] = s;
}

// ---------------------------------------------------------------------------
// Fragment loaders: b128 loads land directly in WMMA register layouts.
//   A (16-bit, 16x32): lanes 0-15: K=0..7 & 16..23; lanes 16-31: K=8..15 & 24..31.
//   B (16-bit, 32x16): lanes 0-15: K=0..7 & 16..23; lanes 16-31: K=8..15 & 24..31.
// ---------------------------------------------------------------------------
__device__ __forceinline__ v16bf load_frag_A(const __bf16* p) {   // chunks at +0, +16
    v8bf c0 = __builtin_bit_cast(v8bf, *(const uint4*)(p));
    v8bf c1 = __builtin_bit_cast(v8bf, *(const uint4*)(p + 16));
    return __builtin_shufflevector(c0, c1, 0,1,2,3,4,5,6,7,8,9,10,11,12,13,14,15);
}

__device__ __forceinline__ v16bf load_frag_B(const __bf16* p) {   // chunks at +0, +16
    v8bf c0 = __builtin_bit_cast(v8bf, *(const uint4*)(p));
    v8bf c1 = __builtin_bit_cast(v8bf, *(const uint4*)(p + 16));
    return __builtin_shufflevector(c0, c1, 0,1,2,3,4,5,6,7,8,9,10,11,12,13,14,15);
}

// ---------------------------------------------------------------------------
// Kernel 3: WMMA GEMM + RBF epilogue.
// Block: 256 threads = 8 waves arranged 2(M) x 4(N); block tile 128 x 256.
// Wave tile: 64 x 64 = 4x4 WMMA 16x16 tiles. Grid: (16, 32).
// ---------------------------------------------------------------------------
__global__ __launch_bounds__(256)
void rbf_wmma_gemm(const __bf16* __restrict__ xhi, const __bf16* __restrict__ xlo,
                   const __bf16* __restrict__ yhi, const __bf16* __restrict__ ylo,
                   const float* __restrict__ gptr,
                   const float* __restrict__ x_sq, const float* __restrict__ y_sq,
                   float* __restrict__ out) {
    const int lane  = (int)(threadIdx.x & 31u);
    const int wave  = (int)(threadIdx.x >> 5);
    const int waveM = wave & 1;   // 0..1
    const int waveN = wave >> 1;  // 0..3

    const int rowBase = (int)blockIdx.y * 128 + waveM * 64;  // 64 rows per wave
    const int colBase = (int)blockIdx.x * 256 + waveN * 64;  // 64 cols per wave

    const int laneLo = lane & 15;
    const int laneHi = lane >> 4;

    v8f c[4][4];
    #pragma unroll
    for (int mt = 0; mt < 4; ++mt)
        #pragma unroll
        for (int nt = 0; nt < 4; ++nt)
            c[mt][nt] = (v8f){0.f,0.f,0.f,0.f,0.f,0.f,0.f,0.f};

    // Loop-invariant per-lane element offsets (32-bit); k0 advances uniformly.
    int aOff[4], bOff[4];
    const int aK0 = laneHi * 8;    // A: two 8-elem chunks at +0, +16 past this
    const int bK0 = laneHi * 8;    // B: two 8-elem chunks at +0, +16 past this
    #pragma unroll
    for (int mt = 0; mt < 4; ++mt)
        aOff[mt] = (rowBase + mt * 16 + laneLo) * F_DIM + aK0;
    #pragma unroll
    for (int nt = 0; nt < 4; ++nt)
        bOff[nt] = (colBase + nt * 16 + laneLo) * F_DIM + bK0;

    for (int k0 = 0; k0 < F_DIM; k0 += 32) {
        // ---- A fragments for all 4 row-tiles (hi + lo) ----
        v16bf aH[4], aL[4];
        #pragma unroll
        for (int mt = 0; mt < 4; ++mt) {
            aH[mt] = load_frag_A(xhi + aOff[mt] + k0);
            aL[mt] = load_frag_A(xlo + aOff[mt] + k0);
        }
        // ---- B fragments in pairs to bound register pressure ----
        #pragma unroll
        for (int np = 0; np < 2; ++np) {
            v16bf bH[2], bL[2];
            #pragma unroll
            for (int j = 0; j < 2; ++j) {
                const int nt = np * 2 + j;
                bH[j] = load_frag_B(yhi + bOff[nt] + k0);
                bL[j] = load_frag_B(ylo + bOff[nt] + k0);
            }
            #pragma unroll
            for (int mt = 0; mt < 4; ++mt) {
                #pragma unroll
                for (int j = 0; j < 2; ++j) {
                    const int nt = np * 2 + j;
                    c[mt][nt] = __builtin_amdgcn_wmma_f32_16x16x32_bf16(
                        false, aH[mt], false, bH[j], (short)0, c[mt][nt], false, false);
                    c[mt][nt] = __builtin_amdgcn_wmma_f32_16x16x32_bf16(
                        false, aH[mt], false, bL[j], (short)0, c[mt][nt], false, false);
                    c[mt][nt] = __builtin_amdgcn_wmma_f32_16x16x32_bf16(
                        false, aL[mt], false, bH[j], (short)0, c[mt][nt], false, false);
                }
            }
        }
    }

    // ---- RBF epilogue (C/D layout: VGPR v -> M = v + 8*laneHi, N = laneLo) ----
    const float gamma = *gptr;

    // Hoist norm lookups: 32 row norms + 4 col norms per lane.
    float xs[4][8], ys[4];
    #pragma unroll
    for (int mt = 0; mt < 4; ++mt)
        #pragma unroll
        for (int v = 0; v < 8; ++v)
            xs[mt][v] = x_sq[rowBase + mt * 16 + v + (laneHi << 3)];
    #pragma unroll
    for (int nt = 0; nt < 4; ++nt)
        ys[nt] = y_sq[colBase + nt * 16 + laneLo];

    #pragma unroll
    for (int mt = 0; mt < 4; ++mt) {
        #pragma unroll
        for (int nt = 0; nt < 4; ++nt) {
            #pragma unroll
            for (int v = 0; v < 8; ++v) {
                const int row = rowBase + mt * 16 + v + (laneHi << 3);
                const int col = colBase + nt * 16 + laneLo;
                float s = xs[mt][v] + ys[nt] - 2.0f * c[mt][nt][v];
                s = fmaxf(s, 0.0f);
                __builtin_nontemporal_store(__expf(-gamma * s),
                                            &out[(size_t)row * B_DIM + col]);
            }
        }
    }
}

// ---------------------------------------------------------------------------
extern "C" void kernel_launch(void* const* d_in, const int* in_sizes, int n_in,
                              void* d_out, int out_size, void* d_ws, size_t ws_size,
                              hipStream_t stream) {
    (void)in_sizes; (void)n_in; (void)out_size; (void)ws_size;
    const float* x     = (const float*)d_in[0];
    const float* y     = (const float*)d_in[1];
    const float* gamma = (const float*)d_in[2];
    float* out = (float*)d_out;

    // Workspace layout (all offsets 256B-aligned):
    //   [0, 32KB)     : sq  (x_sq 4096 floats, y_sq 4096 floats)
    //   [32KB, +8MB)  : xhi   [+8MB) xlo   [+8MB) yhi   [+8MB) ylo
    char* ws = (char*)d_ws;
    float*  sq  = (float*)ws;
    const size_t NEL = (size_t)B_DIM * F_DIM;           // 4M elements
    __bf16* xhi = (__bf16*)(ws + (32u << 10));
    __bf16* xlo = xhi + NEL;
    __bf16* yhi = xlo + NEL;
    __bf16* ylo = yhi + NEL;

    const int n4 = (int)(NEL / 4);                      // 1M float4 per tensor
    bf16_split_kernel<<<dim3(n4 / 256), dim3(256), 0, stream>>>(x, xhi, xlo, n4);
    bf16_split_kernel<<<dim3(n4 / 256), dim3(256), 0, stream>>>(y, yhi, ylo, n4);

    rbf_row_norms<<<dim3(1024), dim3(256), 0, stream>>>(x, y, sq);

    rbf_wmma_gemm<<<dim3(B_DIM / 256, B_DIM / 128), dim3(256), 0, stream>>>(
        xhi, xlo, yhi, ylo, gamma, sq, sq + B_DIM, out);
}
